// FusionLayer_49357764166065
// MI455X (gfx1250) — matrix operand order repaired and checked by reading, not compiled, not run on hardware
//
#include <hip/hip_runtime.h>
#include <math.h>

typedef __attribute__((ext_vector_type(16))) __bf16 v16bf;
typedef __attribute__((ext_vector_type(8)))  float  v8f;

#define C_IN   512
#define C_OUT  256
#define HW     784        // 28*28, divisible by 16 -> M-tiles never cross batches
#define KSTEPS 16         // 512 / 32
#define MTILES 6272       // 100352 / 16

// ---------------------------------------------------------------------------
// Pack fp32 weights W[k][n] (k=0..511, n=0..255, row-major = B-matrix K x N)
// into bf16 hi/lo arrays in a contiguous-k B-fragment order matching the A loads:
//   fragment(n_tile, ks): lane = (k_local>=16)*16 + (n&15) holds 16 bf16
//   (32 contiguous bytes) = K values (lane>=16 ? 16..31 : 0..15) of column n.
// ---------------------------------------------------------------------------
__global__ __launch_bounds__(256) void pack_weights(const float* __restrict__ w,
                                                    __bf16* __restrict__ whi,
                                                    __bf16* __restrict__ wlo) {
    int t = blockIdx.x * 256 + threadIdx.x;
    if (t >= C_IN * C_OUT) return;
    int k = t >> 8;           // 0..511
    int n = t & 255;          // 0..255
    float x = w[t];
    __bf16 hi = (__bf16)x;
    __bf16 lo = (__bf16)(x - (float)hi);
    int lane = (((k & 31) >> 4) << 4) + (n & 15); // half-wave by K, column by N
    int j    = k & 15;                            // position within 32B run
    size_t idx = ((((size_t)(n >> 4) * KSTEPS + (size_t)(k >> 5)) * 32 + lane) * 16 + j);
    whi[idx] = hi;
    wlo[idx] = lo;
}

// ---------------------------------------------------------------------------
// Fused concat + 1x1 conv + bias + tanh as a bf16x3 split-precision WMMA GEMM.
// Workgroup: 256 threads = 8 waves = 2 M-tiles x 4 N-groups (64 cols each).
// Each wave: 16x64 output tile, 4 f32 accumulators, 3 WMMAs per (ks, ntile).
// ---------------------------------------------------------------------------
__global__ __launch_bounds__(256) void fusion_gemm(
    const float* __restrict__ mid,   // [100352, 256] fp32
    const float* __restrict__ glob,  // [128, 256]    fp32
    const float* __restrict__ bias,  // [256]         fp32
    const __bf16* __restrict__ whi,  // packed B hi
    const __bf16* __restrict__ wlo,  // packed B lo
    float* __restrict__ out)         // [100352, 256] fp32
{
    const int lane   = threadIdx.x & 31;
    const int wave   = threadIdx.x >> 5;
    const int m_tile = blockIdx.x * 2 + (wave >> 2);
    const int ngrp   = wave & 3;

    const int m0    = m_tile * 16;
    const int batch = m0 / HW;                 // whole tile shares one batch
    const int row   = m0 + (lane & 15);        // this lane's M row (A layout)

    const float* arow_mid  = mid + (size_t)row * 256;
    const float* arow_glob = glob + (size_t)batch * 256 - 256; // index by k>=256
    const int khalf = (lane >> 4) * 16;        // A: lanes>=16 own K 16..31

    v8f acc[4] = {};                           // 4 N-tiles of 16x16 f32

    const size_t blane = (size_t)lane * 16;

    for (int ks = 0; ks < KSTEPS; ++ks) {
        // ---- load 16 fp32 A elements: run [kb, kb+16) ----
        const float* src = (ks < 8) ? arow_mid : arow_glob;  // uniform branch
        const int kb = ks * 32 + khalf;
        const float4 f0 = *reinterpret_cast<const float4*>(src + kb);
        const float4 f1 = *reinterpret_cast<const float4*>(src + kb + 4);
        const float4 f2 = *reinterpret_cast<const float4*>(src + kb + 8);
        const float4 f3 = *reinterpret_cast<const float4*>(src + kb + 12);
        float fa[16] = { f0.x, f0.y, f0.z, f0.w, f1.x, f1.y, f1.z, f1.w,
                         f2.x, f2.y, f2.z, f2.w, f3.x, f3.y, f3.z, f3.w };

        // ---- split fp32 -> bf16 hi + bf16 lo (VALU, co-executes with WMMA) ----
        v16bf ahi, alo;
#pragma unroll
        for (int e = 0; e < 16; ++e) {
            __bf16 h = (__bf16)fa[e];
            ahi[e] = h;
            alo[e] = (__bf16)(fa[e] - (float)h);
        }

#pragma unroll
        for (int nt = 0; nt < 4; ++nt) {
            const size_t fb = (((size_t)(ngrp * 4 + nt) * KSTEPS + ks) * 32) * 16 + blane;
            v16bf bhi = *reinterpret_cast<const v16bf*>(whi + fb);
            v16bf blo = *reinterpret_cast<const v16bf*>(wlo + fb);
            // hi*hi + hi*lo + lo*hi  ~= full fp32 product, f32 accumulation
            acc[nt] = __builtin_amdgcn_wmma_f32_16x16x32_bf16(
                          false, ahi, false, bhi, (short)0, acc[nt], false, false);
            acc[nt] = __builtin_amdgcn_wmma_f32_16x16x32_bf16(
                          false, ahi, false, blo, (short)0, acc[nt], false, false);
            acc[nt] = __builtin_amdgcn_wmma_f32_16x16x32_bf16(
                          false, alo, false, bhi, (short)0, acc[nt], false, false);
        }
    }

    // ---- epilogue: bias + tanh, D-fragment layout (M = r + 8*(lane>=16)) ----
    const int mrow = m0 + (lane >> 4) * 8;
#pragma unroll
    for (int nt = 0; nt < 4; ++nt) {
        const int n  = ngrp * 64 + nt * 16 + (lane & 15);
        const float bv = bias[n];
#pragma unroll
        for (int r = 0; r < 8; ++r) {
            float v = tanhf(acc[nt][r] + bv);
            __builtin_nontemporal_store(v, out + (size_t)(mrow + r) * 256 + n);
        }
    }
}

extern "C" void kernel_launch(void* const* d_in, const int* in_sizes, int n_in,
                              void* d_out, int out_size, void* d_ws, size_t ws_size,
                              hipStream_t stream) {
    const float* mid  = (const float*)d_in[0];  // [128,28,28,256]
    const float* glob = (const float*)d_in[1];  // [128,256]
    const float* w    = (const float*)d_in[2];  // [1,1,512,256]
    const float* bias = (const float*)d_in[3];  // [256]
    // d_in[4] = stride (==1), SAME padding on a 1x1 conv: identity mapping.

    __bf16* whi = (__bf16*)d_ws;                    // 256 KB
    __bf16* wlo = whi + (size_t)C_IN * C_OUT;       // 256 KB

    pack_weights<<<(C_IN * C_OUT + 255) / 256, 256, 0, stream>>>(w, whi, wlo);
    fusion_gemm<<<MTILES / 2, 256, 0, stream>>>(mid, glob, bias, whi, wlo,
                                                (float*)d_out);
}
